// PTMDialogueEncoder_56375740727798
// MI455X (gfx1250) — compile-verified, matrix operand order checked by reading, not run on hardware
//
#include <hip/hip_runtime.h>
#include <math.h>

// ---------------------------------------------------------------------------
// PTM dialogue encoder for MI455X (gfx1250), wave32 + WMMA f16 (f32 accum).
//
// Roofline: ~11.5 GFLOP of GEMM vs ~20 MB of unavoidable HBM traffic -> the
// problem is matrix-op / L2-bandwidth bound. All GEMMs use
// v_wmma_f32_16x16x32_f16 (8x FLOP/instr vs the f32 16x16x4 WMMA; K=768 with
// f32 accumulation keeps ~1e-3 relative error).
//
// Round-2: hold 4 M-tile A-fragments + 24 accumulators in registers so each
// B fragment feeds 4 WMMAs (64 FLOP/B from L2 instead of 16).
// Round-3: weight fragments are streamed via a single marching base pointer
// with compile-time tile offsets (t*1024 B), so B loads become
// global_load_b128 base+imm instead of per-fragment 64-bit address math;
// this also removes the WMMA->VALU WAR-hazard v_nops that address churn
// forced after each WMMA quad.
// ---------------------------------------------------------------------------

#define B_    64
#define L_    512
#define H_    768
#define N_    50
#define O_    768
#define ALPHA 0.3f

#define KT 24      // 768 / 32 k-steps
#define NT 48      // 768 / 16 col tiles
#define SX 776     // bufX row stride (halfs), 1552B -> 16B aligned rows
#define ST 72      // bufT (Wh^T) row stride (halfs)
#define SA 72      // att row stride (halfs)
#define FRAGH 512  // halfs per 16x32 weight fragment (32 lanes x 16 halfs)

// LDS: bufX[64][SX] + bufT[768][ST] + att[64][SA] (f16) + 5*64 f32 + 64 int
#define SMEM_BYTES ((64 * SX + 768 * ST + 64 * SA) * 2 + 5 * 64 * 4 + 64 * 4)

typedef __attribute__((ext_vector_type(16))) _Float16 v16h;
typedef __attribute__((ext_vector_type(8)))  _Float16 v8h;
typedef __attribute__((ext_vector_type(8)))  float    v8f;

// Load a 16x32 f16 A-fragment (or symmetric B-fragment) from row-major LDS.
// ISA layout (16-bit A 16x32): lane l -> row = row0 + (l&15); halfs 0..7 are
// K = kb + o1 .. +7, halfs 8..15 are K = kb + o1 + 16 .. +7, o1 = (l<16)?0:8.
// Both runs are contiguous -> two ds_load_b128 per fragment.
__device__ __forceinline__ v16h frag_ld_lds(const _Float16* base, int row0,
                                            int kb, int stride, int lane) {
  int r  = row0 + (lane & 15);
  int o1 = (lane & 16) ? 8 : 0;
  const _Float16* p = base + r * stride + kb + o1;
  v8h lo = *(const v8h*)p;
  v8h hi = *(const v8h*)(p + 16);
  return __builtin_shufflevector(lo, hi, 0, 1, 2, 3, 4, 5, 6, 7,
                                 8, 9, 10, 11, 12, 13, 14, 15);
}

__device__ __forceinline__ float elu1(float x) {
  return x > 0.f ? x : __expf(x) - 1.f;
}

// Sinusoidal PE: pe[m][n], pair i = n>>1, freq = exp(-ln(1e4) * 2i / H).
__device__ __forceinline__ float pe_val(int m, int n) {
  float f = __expf(-logf(10000.f) * (float)(n & ~1) / (float)H_);
  float ang = (float)m * f;
  return (n & 1) ? __cosf(ang) : __sinf(ang);
}

// ---------------------------------------------------------------------------
// Prep: convert f32 [768][768] weights (row-major, W[k][n]) into f16
// fragment-major layout: frag[kt*NT+nt][lane][j] so each lane loads one v16h.
// ---------------------------------------------------------------------------
__global__ __launch_bounds__(256) void ptm_prep_kernel(
    const float* W1, const float* W2, const float* Wa, const float* Wl,
    _Float16* W1f, _Float16* W2f, _Float16* Waf, _Float16* Wlf) {
  const float* src;
  _Float16* dst;
  switch (blockIdx.y) {
    case 0:  src = W1; dst = W1f; break;
    case 1:  src = W2; dst = W2f; break;
    case 2:  src = Wa; dst = Waf; break;
    default: src = Wl; dst = Wlf; break;
  }
  int kt = blockIdx.x / NT;
  int nt = blockIdx.x % NT;
  int e0 = threadIdx.x * 2;
#pragma unroll
  for (int u = 0; u < 2; ++u) {
    int e  = e0 + u;
    int l  = e >> 4;       // lane 0..31
    int j  = e & 15;       // half 0..15
    int o1 = (l & 16) ? 8 : 0;
    int k  = kt * 32 + ((j < 8) ? (o1 + j) : (o1 + 16 + (j - 8)));
    int n  = nt * 16 + (l & 15);
    dst[(size_t)blockIdx.x * FRAGH + e] = (_Float16)src[k * H_ + n];
  }
}

// ---------------------------------------------------------------------------
// Main: one block per dialogue. 256 threads = 8 waves; each wave owns 6
// column tiles (nt = wave*6..wave*6+5) and ALL 4 row tiles (M padded 50->64),
// so each B fragment feeds 4 WMMAs.
// ---------------------------------------------------------------------------
__global__ __launch_bounds__(256) void ptm_dialogue_kernel(
    const float* __restrict__ emb, const int* __restrict__ cls,
    const int* __restrict__ adj,
    const _Float16* __restrict__ W1f, const float* __restrict__ b1,
    const float* __restrict__ a1, const float* __restrict__ ab1,
    const _Float16* __restrict__ W2f, const float* __restrict__ b2,
    const float* __restrict__ a2, const float* __restrict__ ab2,
    const _Float16* __restrict__ Waf, const float* __restrict__ ba,
    const float* __restrict__ vvec, float* __restrict__ dial) {
  extern __shared__ unsigned char smem[];
  _Float16* bufX = (_Float16*)smem;            // [64][SX] row-major A input
  _Float16* bufT = bufX + 64 * SX;             // [768][ST] Wh^T (B fragments)
  _Float16* attm = bufT + 768 * ST;            // [64][SA] attention rows
  float* asrc1 = (float*)(attm + 64 * SA);
  float* adst1 = asrc1 + 64;
  float* asrc2 = adst1 + 64;
  float* adst2 = asrc2 + 64;
  float* sc    = adst2 + 64;
  int*   sids  = (int*)(sc + 64);

  const int b    = blockIdx.x;
  const int tid  = threadIdx.x;
  const int lane = tid & 31;
  const int wave = tid >> 5;

  // ---- stage 0: ids, zero-init reductions, gather + positional encoding ----
  if (tid < N_) sids[tid] = cls[b * N_ + tid];
  if (tid < 64) {
    asrc1[tid] = 0.f; adst1[tid] = 0.f;
    asrc2[tid] = 0.f; adst2[tid] = 0.f;
    sc[tid] = 0.f;
  }
  __syncthreads();
  for (int idx = tid; idx < 64 * H_; idx += 256) {
    int m = idx / H_;
    int n = idx - m * H_;
    float x = 0.f;
    if (m < N_) x = emb[((size_t)b * L_ + sids[m]) * H_ + n] + pe_val(m, n);
    bufX[m * SX + n] = (_Float16)x;
  }
  __syncthreads();

  // ---- two GAT layers ----
  for (int layer = 0; layer < 2; ++layer) {
    const _Float16* Wf = layer ? W2f : W1f;
    const float* bb = layer ? b2 : b1;
    const float* aa = layer ? a2 : a1;
    float abv = layer ? ab2[0] : ab1[0];
    float* asrc = layer ? asrc2 : asrc1;
    float* adst = layer ? adst2 : adst1;

    // Wh = bufX @ W + b  ->  bufT = Wh^T (f16). B fragment reused 4x.
    {
      v8f acc[4][6] = {};
      // marching fragment pointer: wave's 6 tiles at constant offsets t*FRAGH
      const _Float16* wp = Wf + (size_t)(wave * 6) * FRAGH + lane * 16;
      for (int ks = 0; ks < KT; ++ks, wp += (size_t)NT * FRAGH) {
        v16h a[4];
#pragma unroll
        for (int mt = 0; mt < 4; ++mt)
          a[mt] = frag_ld_lds(bufX, mt * 16, ks * 32, SX, lane);
#pragma unroll
        for (int t = 0; t < 6; ++t) {
          v16h bf = *(const v16h*)(wp + t * FRAGH);
#pragma unroll
          for (int mt = 0; mt < 4; ++mt)
            acc[mt][t] = __builtin_amdgcn_wmma_f32_16x16x32_f16(
                false, a[mt], false, bf, (short)0, acc[mt][t], false, false);
        }
      }
      int hi = (lane & 16) ? 8 : 0;
#pragma unroll
      for (int t = 0; t < 6; ++t) {
        int n = (wave * 6 + t) * 16 + (lane & 15);
        float bias = bb[n];
#pragma unroll
        for (int mt = 0; mt < 4; ++mt) {
          v8h pack;
#pragma unroll
          for (int r = 0; r < 8; ++r)
            pack[r] = (_Float16)(acc[mt][t][r] + bias);
          // rows m = mt*16 + hi + r are contiguous -> one 16B LDS store
          *(v8h*)(bufT + n * ST + mt * 16 + hi) = pack;
        }
      }
    }
    __syncthreads();

    // a_src[m] = Wh[m]·a[:H], a_dst[m] = Wh[m]·a[H:]  (from Wh^T columns)
    {
      int m = tid >> 2;
      int q = tid & 3;
      float s = 0.f, d = 0.f;
      for (int n = q * 192; n < q * 192 + 192; ++n) {
        float w = (float)bufT[n * ST + m];
        s += w * aa[n];
        d += w * aa[H_ + n];
      }
      atomicAdd(&asrc[m], s);
      atomicAdd(&adst[m], d);
    }
    __syncthreads();

    // masked leaky-relu softmax rows -> attm (f16), padded rows/cols zeroed
    if (tid < 64) {
      int m = tid;
      if (m < N_) {
        float am = asrc[m] + abv;
        const int* arow = adj + ((size_t)b * N_ + m) * N_;
        float mx = -3.0e38f;
        for (int n = 0; n < N_; ++n) {
          if (arow[n] > 0) {
            float e = am + adst[n];
            e = (e >= 0.f) ? e : ALPHA * e;
            mx = fmaxf(mx, e);
          }
        }
        float ssum = 0.f;
        for (int n = 0; n < N_; ++n) {
          if (arow[n] > 0) {
            float e = am + adst[n];
            e = (e >= 0.f) ? e : ALPHA * e;
            ssum += __expf(e - mx);
          }
        }
        float inv = 1.f / ssum;
        for (int n = 0; n < N_; ++n) {
          float w = 0.f;
          if (arow[n] > 0) {
            float e = am + adst[n];
            e = (e >= 0.f) ? e : ALPHA * e;
            w = __expf(e - mx) * inv;
          }
          attm[m * SA + n] = (_Float16)w;
        }
        for (int n = N_; n < 64; ++n) attm[m * SA + n] = (_Float16)0.f;
      } else {
        for (int n = 0; n < 64; ++n) attm[m * SA + n] = (_Float16)0.f;
      }
    }
    __syncthreads();

    // h = elu(att @ Wh) (+ x residual on layer 2) -> bufX (row-major f16)
    {
      v8f acc[4][6] = {};
#pragma unroll
      for (int ks = 0; ks < 2; ++ks) {
        v16h a[4];
#pragma unroll
        for (int mt = 0; mt < 4; ++mt)
          a[mt] = frag_ld_lds(attm, mt * 16, ks * 32, SA, lane);
#pragma unroll
        for (int t = 0; t < 6; ++t) {
          v16h bf = frag_ld_lds(bufT, (wave * 6 + t) * 16, ks * 32, ST, lane);
#pragma unroll
          for (int mt = 0; mt < 4; ++mt)
            acc[mt][t] = __builtin_amdgcn_wmma_f32_16x16x32_f16(
                false, a[mt], false, bf, (short)0, acc[mt][t], false, false);
        }
      }
      int hi = (lane & 16) ? 8 : 0;
#pragma unroll
      for (int t = 0; t < 6; ++t) {
        int n = (wave * 6 + t) * 16 + (lane & 15);
#pragma unroll
        for (int mt = 0; mt < 4; ++mt) {
#pragma unroll
          for (int r = 0; r < 8; ++r) {
            int m = mt * 16 + hi + r;
            float hv = 0.f;
            if (m < N_) {
              hv = elu1(acc[mt][t][r]);
              if (layer == 1)  // residual: recompute gathered x + PE
                hv += emb[((size_t)b * L_ + sids[m]) * H_ + n] + pe_val(m, n);
            }
            bufX[m * SX + n] = (_Float16)hv;
          }
        }
      }
    }
    __syncthreads();
  }

  // ---- attention pooling: sc[m] = sum_n tanh(h@Wa + ba)[m][n] * v[n] ----
  {
    v8f acc[4][6] = {};
    const _Float16* wp = Waf + (size_t)(wave * 6) * FRAGH + lane * 16;
    for (int ks = 0; ks < KT; ++ks, wp += (size_t)NT * FRAGH) {
      v16h a[4];
#pragma unroll
      for (int mt = 0; mt < 4; ++mt)
        a[mt] = frag_ld_lds(bufX, mt * 16, ks * 32, SX, lane);
#pragma unroll
      for (int t = 0; t < 6; ++t) {
        v16h bf = *(const v16h*)(wp + t * FRAGH);
#pragma unroll
        for (int mt = 0; mt < 4; ++mt)
          acc[mt][t] = __builtin_amdgcn_wmma_f32_16x16x32_f16(
              false, a[mt], false, bf, (short)0, acc[mt][t], false, false);
      }
    }
    int hi = (lane & 16) ? 8 : 0;
#pragma unroll
    for (int t = 0; t < 6; ++t) {
      int n = (wave * 6 + t) * 16 + (lane & 15);
      float bav = ba[n];
      float vv = vvec[n];
#pragma unroll
      for (int mt = 0; mt < 4; ++mt) {
#pragma unroll
        for (int r = 0; r < 8; ++r) {
          int m = mt * 16 + hi + r;
          if (m < N_) atomicAdd(&sc[m], tanhf(acc[mt][t][r] + bav) * vv);
        }
      }
    }
  }
  __syncthreads();

  // softmax over the 50 scores
  if (tid == 0) {
    float mx = -3.0e38f;
    for (int m = 0; m < N_; ++m) mx = fmaxf(mx, sc[m]);
    float s = 0.f;
    for (int m = 0; m < N_; ++m) {
      float e = __expf(sc[m] - mx);
      sc[m] = e;
      s += e;
    }
    float inv = 1.f / s;
    for (int m = 0; m < N_; ++m) sc[m] *= inv;
  }
  __syncthreads();

  // dialogue vector: dial[b][n] = sum_m sc[m] * h[m][n]
  for (int n = tid; n < H_; n += 256) {
    float s = 0.f;
    for (int m = 0; m < N_; ++m) s += sc[m] * (float)bufX[m * SX + n];
    dial[(size_t)b * H_ + n] = s;
  }
}

// ---------------------------------------------------------------------------
// Final: out[64][768] = dial @ Wl + bl. 48 blocks (col tiles) x 4 waves (row
// tiles); A fragments converted from f32 on the fly.
// ---------------------------------------------------------------------------
__global__ __launch_bounds__(128) void ptm_final_kernel(
    const float* __restrict__ dial, const _Float16* __restrict__ Wlf,
    const float* __restrict__ bl, float* __restrict__ out) {
  int lane = threadIdx.x & 31;
  int mt = threadIdx.x >> 5;  // 0..3
  int nt = blockIdx.x;        // 0..47
  int o1 = (lane & 16) ? 8 : 0;
  int m = mt * 16 + (lane & 15);
  const float* arow = dial + (size_t)m * H_;
  const _Float16* wp = Wlf + (size_t)nt * FRAGH + lane * 16;
  v8f acc = {};
  for (int ks = 0; ks < KT; ++ks, wp += (size_t)NT * FRAGH) {
    int kb = ks * 32;
    v16h a;
#pragma unroll
    for (int i = 0; i < 8; ++i) a[i] = (_Float16)arow[kb + o1 + i];
#pragma unroll
    for (int i = 0; i < 8; ++i) a[8 + i] = (_Float16)arow[kb + o1 + 16 + i];
    v16h bf = *(const v16h*)wp;
    acc = __builtin_amdgcn_wmma_f32_16x16x32_f16(false, a, false, bf, (short)0,
                                                 acc, false, false);
  }
  int n = nt * 16 + (lane & 15);
  int hi = (lane & 16) ? 8 : 0;
  float blv = bl[n];
#pragma unroll
  for (int r = 0; r < 8; ++r) {
    int mr = mt * 16 + hi + r;
    out[(size_t)mr * O_ + n] = acc[r] + blv;
  }
}

extern "C" void kernel_launch(void* const* d_in, const int* in_sizes, int n_in,
                              void* d_out, int out_size, void* d_ws,
                              size_t ws_size, hipStream_t stream) {
  const float* emb = (const float*)d_in[0];
  const int* cls   = (const int*)d_in[1];
  const int* adj   = (const int*)d_in[2];
  const float* W1  = (const float*)d_in[3];
  const float* b1  = (const float*)d_in[4];
  const float* a1  = (const float*)d_in[5];
  const float* ab1 = (const float*)d_in[6];
  const float* W2  = (const float*)d_in[7];
  const float* b2  = (const float*)d_in[8];
  const float* a2  = (const float*)d_in[9];
  const float* ab2 = (const float*)d_in[10];
  const float* Wa  = (const float*)d_in[11];
  const float* ba  = (const float*)d_in[12];
  const float* v   = (const float*)d_in[13];
  const float* Wl  = (const float*)d_in[14];
  const float* bl  = (const float*)d_in[15];

  // workspace: 4 x f16 fragment weights (1.125 MB each) + dial [64][768] f32
  const size_t FRAG = (size_t)KT * NT * FRAGH;  // halfs per weight matrix
  _Float16* W1f = (_Float16*)d_ws;
  _Float16* W2f = W1f + FRAG;
  _Float16* Waf = W2f + FRAG;
  _Float16* Wlf = Waf + FRAG;
  float* dial   = (float*)(Wlf + FRAG);

  // 215 KB dynamic LDS per workgroup (CDNA5 allows up to 320 KB per WG)
  (void)hipFuncSetAttribute((const void*)ptm_dialogue_kernel,
                            hipFuncAttributeMaxDynamicSharedMemorySize,
                            SMEM_BYTES);

  ptm_prep_kernel<<<dim3(KT * NT, 4), 256, 0, stream>>>(W1, W2, Wa, Wl, W1f,
                                                        W2f, Waf, Wlf);
  ptm_dialogue_kernel<<<dim3(B_), 256, SMEM_BYTES, stream>>>(
      emb, cls, adj, W1f, b1, a1, ab1, W2f, b2, a2, ab2, Waf, ba, v, dial);
  ptm_final_kernel<<<dim3(NT), 128, 0, stream>>>(dial, Wlf, bl, (float*)d_out);
}
